// SelfAttention_46789373723386
// MI455X (gfx1250) — matrix operand order repaired and checked by reading, not compiled, not run on hardware
//
#include <hip/hip_runtime.h>

#define B_  512
#define T_  256
#define C_  768
#define H_  64

typedef __attribute__((ext_vector_type(16))) _Float16 v16h;
typedef __attribute__((ext_vector_type(8)))  _Float16 v8h;
typedef __attribute__((ext_vector_type(8)))  float    v8f;
typedef __attribute__((ext_vector_type(4)))  float    v4f;

static __device__ __forceinline__ v8f wmma_f16(v16h a, v16h b, v8f c) {
  // D = A(16x32 f16) * B(32x16 f16) + C(16x16 f32)
  return __builtin_amdgcn_wmma_f32_16x16x32_f16(false, a, false, b, (short)0, c,
                                                false, false);
}

static __device__ __forceinline__ v16h cat8(v8h lo, v8h hi) {
  return __builtin_shufflevector(lo, hi, 0, 1, 2, 3, 4, 5, 6, 7,
                                 8, 9, 10, 11, 12, 13, 14, 15);
}

// ---------------------------------------------------------------------------
// Kernel 1: W[C,H] fp32 -> WT[3][H][C] f16 (transposed so B-operand loads are
// contiguous per lane).
// ---------------------------------------------------------------------------
__global__ void wt_kernel(const float* __restrict__ Wq,
                          const float* __restrict__ Wk,
                          const float* __restrict__ Wv,
                          _Float16* __restrict__ WT) {
  int i = blockIdx.x * blockDim.x + threadIdx.x;
  const int n = 3 * H_ * C_;
  if (i >= n) return;
  int w   = i / (H_ * C_);
  int rem = i - w * (H_ * C_);
  int h   = rem / C_;
  int c   = rem - h * C_;
  const float* W = (w == 0) ? Wq : (w == 1) ? Wk : Wv;
  WT[i] = (_Float16)W[c * H_ + h];
}

// ---------------------------------------------------------------------------
// Kernel 2: projections. One wave per 16-row tile of x.
// Computes Q,K -> [B,T,H] f16 and V -> VT[B,H,T] f16.
// 12 live accumulators (Q,K,V x 4 n-tiles), K-loop over C in steps of 32.
// ---------------------------------------------------------------------------
__global__ __launch_bounds__(128) void proj_kernel(
    const float* __restrict__ x, const _Float16* __restrict__ WT,
    _Float16* __restrict__ Qm, _Float16* __restrict__ Km,
    _Float16* __restrict__ VTm) {
  const int lane = threadIdx.x & 31;
  const int wave = (blockIdx.x * blockDim.x + threadIdx.x) >> 5;
  const int b    = wave >> 4;          // 16 tiles per batch (T/16)
  const int t0   = (wave & 15) << 4;
  const int ln   = lane & 15;
  const int lg   = lane >> 4;
  const int kbA  = lg * 16;            // A-layout K base
  const int kbB  = lg * 16;            // B-layout K base

  const float* xrow = x + ((size_t)(b * T_ + t0 + ln)) * C_;

  v8f acc[12];
#pragma unroll
  for (int i = 0; i < 12; ++i) acc[i] = {};

  for (int kk = 0; kk < C_; kk += 32) {
    // --- A tile: x rows, f32 -> f16 (A layout: K = kbA+{0..15})
    const float* pa = xrow + kk + kbA;
    v4f f0 = *(const v4f*)(pa + 0);
    v4f f1 = *(const v4f*)(pa + 4);
    v4f f2 = *(const v4f*)(pa + 8);
    v4f f3 = *(const v4f*)(pa + 12);
    v16h a;
#pragma unroll
    for (int j = 0; j < 4; ++j) {
      a[j]      = (_Float16)f0[j];
      a[4 + j]  = (_Float16)f1[j];
      a[8 + j]  = (_Float16)f2[j];
      a[12 + j] = (_Float16)f3[j];
    }
    // --- 3 matrices x 4 n-tiles, reusing A
#pragma unroll
    for (int w = 0; w < 3; ++w) {
#pragma unroll
      for (int nt = 0; nt < 4; ++nt) {
        const _Float16* bp =
            WT + ((size_t)w * H_ + nt * 16 + ln) * C_ + kk + kbB;
        v16h bv = *(const v16h*)bp;  // contiguous 16 halfs (B layout)
        acc[w * 4 + nt] = wmma_f16(a, bv, acc[w * 4 + nt]);
      }
    }
  }

  // --- epilogue: C/D layout -> memory
#pragma unroll
  for (int nt = 0; nt < 4; ++nt) {
    v8f aq = acc[0 * 4 + nt];
    v8f ak = acc[1 * 4 + nt];
    v8f av = acc[2 * 4 + nt];
    const int h = nt * 16 + ln;
#pragma unroll
    for (int r = 0; r < 8; ++r) {
      int t = t0 + r + 8 * lg;
      size_t idx = ((size_t)(b * T_ + t)) * H_ + h;
      Qm[idx] = (_Float16)aq[r];
      Km[idx] = (_Float16)ak[r];
    }
    // VT[b][h][t]: t contiguous per lane -> one 16B store
    v8h vh;
#pragma unroll
    for (int r = 0; r < 8; ++r) vh[r] = (_Float16)av[r];
    _Float16* vp = VTm + ((size_t)b * H_ + h) * T_ + t0 + 8 * lg;
    *(v8h*)vp = vh;
  }
}

// ---------------------------------------------------------------------------
// Kernel 3: flash attention. One wave per 16-query tile; key chunks of 32.
// ---------------------------------------------------------------------------
__global__ __launch_bounds__(128) void attn_kernel(
    const _Float16* __restrict__ Qm, const _Float16* __restrict__ Km,
    const _Float16* __restrict__ VTm, const int* __restrict__ pad,
    float* __restrict__ out) {
  __shared__ _Float16 Plds[4][16 * 32];  // per-wave P bounce buffer

  const int lane = threadIdx.x & 31;
  const int wid  = threadIdx.x >> 5;
  const int wave = blockIdx.x * 4 + wid;
  const int b    = wave >> 4;
  const int t0   = (wave & 15) << 4;
  const int ln   = lane & 15;
  const int lg   = lane >> 4;
  const int kbA  = lg * 16;
  const int kbB  = lg * 16;
  _Float16* P = &Plds[wid][0];

  // --- load Q A-tiles once (two K-steps over H=64), kept in registers
  const _Float16* qrow = Qm + ((size_t)(b * T_ + t0 + ln)) * H_;
  v16h aq0 = cat8(*(const v8h*)(qrow + kbA), *(const v8h*)(qrow + kbA + 8));
  v16h aq1 = cat8(*(const v8h*)(qrow + 32 + kbA),
                  *(const v8h*)(qrow + 32 + kbA + 8));

  v8f O[4];
#pragma unroll
  for (int i = 0; i < 4; ++i) O[i] = {};
  float mrun[8], lrun[8];
#pragma unroll
  for (int r = 0; r < 8; ++r) { mrun[r] = -3.0e38f; lrun[r] = 0.0f; }

  const float scale = 0.125f;  // 1/sqrt(64)
  const int nch = (t0 + 16 + 31) >> 5;

  for (int ch = 0; ch < nch; ++ch) {
    const int s0 = ch * 32;

    // --- scores S[16 x 32] = Q @ K^T
    v8f S[2];
#pragma unroll
    for (int c = 0; c < 2; ++c) {
      S[c] = {};
      const _Float16* krow =
          Km + ((size_t)(b * T_ + s0 + c * 16 + ln)) * H_;
      v16h bk0 = *(const v16h*)(krow + kbB);       // h = kbB..kbB+15
      v16h bk1 = *(const v16h*)(krow + 32 + kbB);  // h = 32+kbB..
      S[c] = wmma_f16(aq0, bk0, S[c]);
      S[c] = wmma_f16(aq1, bk1, S[c]);
    }

    // --- causal mask + online softmax (row r+8*lg lives in VGPR slot r,
    //     spread across a 16-lane group -> butterfly reduce with shfl_xor)
#pragma unroll
    for (int r = 0; r < 8; ++r) {
      const int m  = t0 + r + 8 * lg;  // global query index
      const int sA = s0 + ln;
      const int sB = s0 + 16 + ln;
      float v0 = S[0][r] * scale; if (sA > m) v0 = -1.0e9f;
      float v1 = S[1][r] * scale; if (sB > m) v1 = -1.0e9f;
      float mx = fmaxf(v0, v1);
#pragma unroll
      for (int off = 1; off < 16; off <<= 1)
        mx = fmaxf(mx, __shfl_xor(mx, off, 32));
      const float mnew  = fmaxf(mrun[r], mx);
      const float alpha = __expf(mrun[r] - mnew);
      const float p0 = __expf(v0 - mnew);
      const float p1 = __expf(v1 - mnew);
      S[0][r] = p0;
      S[1][r] = p1;
      float rs = p0 + p1;
#pragma unroll
      for (int off = 1; off < 16; off <<= 1)
        rs += __shfl_xor(rs, off, 32);
      lrun[r] = lrun[r] * alpha + rs;
      mrun[r] = mnew;
      O[0][r] *= alpha; O[1][r] *= alpha; O[2][r] *= alpha; O[3][r] *= alpha;
    }

    // --- P: accumulator layout -> LDS -> A layout (wave-local, compiler
    //     inserts the ds waits)
#pragma unroll
    for (int r = 0; r < 8; ++r) {
      const int m = r + 8 * lg;
      P[m * 32 + ln]      = (_Float16)S[0][r];
      P[m * 32 + 16 + ln] = (_Float16)S[1][r];
    }
    const _Float16* pr = P + ln * 32 + kbA;
    v16h ap = cat8(*(const v8h*)pr, *(const v8h*)(pr + 8));

    // --- O += P @ V  (VT rows make the B-operand contiguous per lane)
#pragma unroll
    for (int nt = 0; nt < 4; ++nt) {
      const _Float16* vrow =
          VTm + ((size_t)(b * H_ + nt * 16 + ln)) * T_ + s0 + kbB;
      v16h bv = *(const v16h*)vrow;
      O[nt] = wmma_f16(ap, bv, O[nt]);
    }
  }

  // --- epilogue: normalize, zero padded query rows, store fp32
#pragma unroll
  for (int r = 0; r < 8; ++r) {
    const int m = t0 + r + 8 * lg;
    const float inv =
        (pad[b * T_ + m] != 0) ? (1.0f / lrun[r]) : 0.0f;
#pragma unroll
    for (int nt = 0; nt < 4; ++nt) {
      out[((size_t)(b * T_ + m)) * H_ + nt * 16 + ln] = O[nt][r] * inv;
    }
  }
}

// ---------------------------------------------------------------------------
extern "C" void kernel_launch(void* const* d_in, const int* in_sizes, int n_in,
                              void* d_out, int out_size, void* d_ws,
                              size_t ws_size, hipStream_t stream) {
  const float* x  = (const float*)d_in[0];
  const float* Wq = (const float*)d_in[1];
  const float* Wk = (const float*)d_in[2];
  const float* Wv = (const float*)d_in[3];
  const int* pad  = (const int*)d_in[4];
  float* out = (float*)d_out;

  char* ws = (char*)d_ws;
  _Float16* WT = (_Float16*)ws;  // [3][H][C]
  size_t off = (size_t)3 * H_ * C_ * sizeof(_Float16);
  off = (off + 255) & ~(size_t)255;
  const size_t qkv_bytes = (size_t)B_ * T_ * H_ * sizeof(_Float16);
  _Float16* Qm  = (_Float16*)(ws + off); off += qkv_bytes;
  _Float16* Km  = (_Float16*)(ws + off); off += qkv_bytes;
  _Float16* VTm = (_Float16*)(ws + off);

  wt_kernel<<<(3 * H_ * C_ + 255) / 256, 256, 0, stream>>>(Wq, Wk, Wv, WT);

  const int tiles = B_ * T_ / 16;  // 8192 waves, 4 waves per block
  proj_kernel<<<tiles / 4, 128, 0, stream>>>(x, WT, Qm, Km, VTm);
  attn_kernel<<<tiles / 4, 128, 0, stream>>>(Qm, Km, VTm, pad, out);
}
